// AttentionProj_7516192768833
// MI455X (gfx1250) — matrix operand order repaired and checked by reading, not compiled, not run on hardware
//
#include <hip/hip_runtime.h>
#include <hip/hip_bf16.h>
#include <math.h>

typedef __attribute__((ext_vector_type(16))) _Float16 v16h;
typedef __attribute__((ext_vector_type(8)))  _Float16 v8h;
typedef __attribute__((ext_vector_type(8)))  float    v8f;
typedef __attribute__((ext_vector_type(4)))  unsigned int u32x4;
typedef __attribute__((ext_vector_type(8)))  int      i32x8;
typedef __attribute__((ext_vector_type(4)))  int      i32x4;

#define N_  64
#define L_  2048
#define HS_ 1024
#define E_  512
#define A_  512

// ---------------------------------------------------------------------------
// qe[n,e] = bq[e] + dot(h_dec[n,:], Wq[e,:])   (tiny: 64x512, K=1024)
// ---------------------------------------------------------------------------
__global__ __launch_bounds__(256) void k_qe(const float* __restrict__ h_dec,
                                            const float* __restrict__ Wq,
                                            const float* __restrict__ bq,
                                            float* __restrict__ qe) {
    int idx = blockIdx.x * 256 + threadIdx.x;       // N_*E_ threads
    int n = idx >> 9, e = idx & (E_ - 1);
    const float4* hp = (const float4*)(h_dec + (size_t)n * HS_);
    const float4* wp = (const float4*)(Wq + (size_t)e * HS_);
    float acc = bq[e];
    for (int k = 0; k < HS_ / 4; ++k) {
        float4 h = hp[k], w = wp[k];
        acc += h.x * w.x + h.y * w.y + h.z * w.z + h.w * w.w;
    }
    qe[idx] = acc;
}

// ---------------------------------------------------------------------------
// qh[n,a] = bh[a] + dot(qe[n,:], Wh[a, 0:E])    (Wh row stride 2E)
// ---------------------------------------------------------------------------
__global__ __launch_bounds__(256) void k_qh(const float* __restrict__ qe,
                                            const float* __restrict__ Wh,
                                            const float* __restrict__ bh,
                                            float* __restrict__ qh) {
    int idx = blockIdx.x * 256 + threadIdx.x;       // N_*A_ threads
    int n = idx >> 9, a = idx & (A_ - 1);
    const float4* qp = (const float4*)(qe + (size_t)n * E_);
    const float4* wp = (const float4*)(Wh + (size_t)a * (2 * E_));
    float acc = bh[a];
    for (int k = 0; k < E_ / 4; ++k) {
        float4 q = qp[k], w = wp[k];
        acc += q.x * w.x + q.y * w.y + q.z * w.z + q.w * w.w;
    }
    qh[idx] = acc;
}

// ---------------------------------------------------------------------------
// TDM helper: issue tensor_load_to_lds of a 2D tile (rows x 32 f32, row
// stride HS_) from global into LDS at lds_off. Built per CDNA5 D# layout:
//   group0: count=1 | lds_addr | global_addr[56:0] | type=2
//   group1: data_size=4B, tensor_dim0=1024, tensor_dim1=rows,
//           tile_dim0=32, tile_dim1=rows, tensor_dim0_stride=1024
// ---------------------------------------------------------------------------
__device__ __forceinline__ void tdm_load_tile(const float* gsrc,
                                              unsigned int lds_off,
                                              unsigned int rows) {
    unsigned long long ga = (unsigned long long)(uintptr_t)gsrc;
    u32x4 g0;
    g0[0] = 1u;                                            // count=1
    g0[1] = lds_off;                                       // lds_addr (bytes)
    g0[2] = (unsigned int)ga;                              // global_addr[31:0]
    g0[3] = ((unsigned int)(ga >> 32) & 0x01FFFFFFu) | (2u << 30); // addr[56:32]|type=2

    i32x8 g1;
    g1[0] = (int)(2u << 16);                 // data_size = 4 bytes
    g1[1] = (int)(1024u << 16);              // tensor_dim0 = 1024 (lo16 here)
    g1[2] = (int)(rows << 16);               // dim0 hi=0 | tensor_dim1 = rows
    g1[3] = (int)(32u << 16);                // dim1 hi=0 | tile_dim0 = 32
    g1[4] = (int)rows;                       // tile_dim1 = rows, tile_dim2 = 0
    g1[5] = 1024;                            // tensor_dim0_stride = 1024
    g1[6] = 0;                               // stride hi / dim1_stride lo
    g1[7] = 0;

    i32x4 gz = {0, 0, 0, 0};
#if __clang_major__ >= 23
    i32x8 gz8 = {0, 0, 0, 0, 0, 0, 0, 0};
    __builtin_amdgcn_tensor_load_to_lds(g0, g1, gz, gz, gz8, 0);
#else
    __builtin_amdgcn_tensor_load_to_lds(g0, g1, gz, gz, 0);
#endif
}

// ---------------------------------------------------------------------------
// ke[nl,e] = f16( bk[e] + sum_k h_enc[nl,k] * Wk[e,k] )
// Block = 8 waves, covers 64 rows (4 M-tiles) x all 512 e-cols.
// A tile (64x32 f32) staged into LDS by the Tensor Data Mover, double
// buffered (TENSORcnt + barrier). Each wave owns 4 e-tiles; B-fragments are
// reused across the 4 M-tiles to cut Wk L2 traffic 4x.
// ---------------------------------------------------------------------------
__global__ __launch_bounds__(256) void k_ke(const float* __restrict__ h_enc,
                                            const float* __restrict__ Wk,
                                            const float* __restrict__ bk,
                                            _Float16* __restrict__ ke) {
    __shared__ float Abuf[2][64 * 32];             // 2 x 8 KB

    const int wave  = threadIdx.x >> 5;
    const int lane  = threadIdx.x & 31;
    const int half  = lane >> 4;                   // K-half selector
    const int lrow  = lane & 15;                   // A: row, B/D: column
    const size_t m0 = (size_t)blockIdx.x * 64;     // 64 rows per block

    v8f acc[4][4] = {};                            // [m-tile][e-tile]
    const float* brow[4];
    float bkv[4];
#pragma unroll
    for (int t = 0; t < 4; ++t) {
        int e = (wave * 4 + t) * 16 + lrow;
        brow[t] = Wk + (size_t)e * HS_;
        bkv[t]  = bk[e];
    }

    // Prologue: TDM-stage first A tile
    if (wave == 0) {
        tdm_load_tile(h_enc + m0 * HS_, (unsigned int)(uintptr_t)&Abuf[0][0], 64u);
        __builtin_amdgcn_s_wait_tensorcnt(0);
    }
    __syncthreads();

    int buf = 0;
    for (int k0 = 0; k0 < HS_; k0 += 32) {
        // Prefetch next A tile into the other buffer while computing
        if (wave == 0 && k0 + 32 < HS_) {
            tdm_load_tile(h_enc + m0 * HS_ + (k0 + 32),
                          (unsigned int)(uintptr_t)&Abuf[buf ^ 1][0], 64u);
        }

        // A fragments from LDS: lane half selects K {0..7,16..23} vs {8..15,24..31}
        v16h aF[4];
#pragma unroll
        for (int m = 0; m < 4; ++m) {
            const float* ab = &Abuf[buf][(m * 16 + lrow) * 32 + half * 8];
            const float4 f0 = *(const float4*)(ab);
            const float4 f1 = *(const float4*)(ab + 4);
            const float4 f2 = *(const float4*)(ab + 16);
            const float4 f3 = *(const float4*)(ab + 20);
            v16h a;
            a[0] = (_Float16)f0.x; a[1] = (_Float16)f0.y; a[2] = (_Float16)f0.z; a[3] = (_Float16)f0.w;
            a[4] = (_Float16)f1.x; a[5] = (_Float16)f1.y; a[6] = (_Float16)f1.z; a[7] = (_Float16)f1.w;
            a[8] = (_Float16)f2.x; a[9] = (_Float16)f2.y; a[10] = (_Float16)f2.z; a[11] = (_Float16)f2.w;
            a[12] = (_Float16)f3.x; a[13] = (_Float16)f3.y; a[14] = (_Float16)f3.z; a[15] = (_Float16)f3.w;
            aF[m] = a;
        }

#pragma unroll
        for (int t = 0; t < 4; ++t) {
            // B fragment: lane n holds column n, K = k0 + half*8 + {0..7, 16..23}
            v16h b;
            const float4 g0 = *(const float4*)(brow[t] + k0 + half * 8);
            const float4 g1 = *(const float4*)(brow[t] + k0 + half * 8 + 4);
            const float4 g2 = *(const float4*)(brow[t] + k0 + half * 8 + 16);
            const float4 g3 = *(const float4*)(brow[t] + k0 + half * 8 + 20);
            b[0] = (_Float16)g0.x; b[1] = (_Float16)g0.y; b[2] = (_Float16)g0.z; b[3] = (_Float16)g0.w;
            b[4] = (_Float16)g1.x; b[5] = (_Float16)g1.y; b[6] = (_Float16)g1.z; b[7] = (_Float16)g1.w;
            b[8] = (_Float16)g2.x; b[9] = (_Float16)g2.y; b[10] = (_Float16)g2.z; b[11] = (_Float16)g2.w;
            b[12] = (_Float16)g3.x; b[13] = (_Float16)g3.y; b[14] = (_Float16)g3.z; b[15] = (_Float16)g3.w;
#pragma unroll
            for (int m = 0; m < 4; ++m) {
                acc[m][t] = __builtin_amdgcn_wmma_f32_16x16x32_f16(
                    false, aF[m], false, b, (short)0, acc[m][t], false, false);
            }
        }

        __syncthreads();                           // everyone done with Abuf[buf]
        if (wave == 0) __builtin_amdgcn_s_wait_tensorcnt(0);
        __syncthreads();                           // next tile visible to all
        buf ^= 1;
    }

    // D layout: lane col = lrow, vgpr r = row r + half*8
#pragma unroll
    for (int m = 0; m < 4; ++m) {
#pragma unroll
        for (int t = 0; t < 4; ++t) {
            int e = (wave * 4 + t) * 16 + lrow;
#pragma unroll
            for (int r = 0; r < 8; ++r) {
                size_t row = m0 + m * 16 + r + half * 8;
                ke[row * E_ + e] = (_Float16)(acc[m][t][r] + bkv[t]);
            }
        }
    }
}

// ---------------------------------------------------------------------------
// score[nl] = sum_a v[a] * tanh( qh[n,a] + sum_e ke[nl,e] * Wh[a, E+e] )
// WMMA over K=E=512 reading f16 ke; fused tanh + v-dot epilogue; wave
// shfl_xor reduction + LDS cross-wave reduction.
// ---------------------------------------------------------------------------
__global__ __launch_bounds__(256) void k_score(const _Float16* __restrict__ ke,
                                               const float* __restrict__ Wh,
                                               const float* __restrict__ qh,
                                               const float* __restrict__ v,
                                               float* __restrict__ score) {
    __shared__ float red[8][16];
    const int mtile = blockIdx.x;
    const int wave  = threadIdx.x >> 5;
    const int lane  = threadIdx.x & 31;
    const int half  = lane >> 4;
    const int lrow  = lane & 15;
    const size_t m0 = (size_t)mtile * 16;
    const int n     = (int)(m0 / L_);

    const _Float16* arow = ke + (m0 + lrow) * E_;

    v8f acc[4] = {};
    const float* brow[4];
    float qv[4], vv[4];
    for (int t = 0; t < 4; ++t) {
        int a0 = (wave * 4 + t) * 16 + lrow;
        brow[t] = Wh + (size_t)a0 * (2 * E_) + E_;   // Wh_k row
        qv[t]   = qh[n * A_ + a0];                    // includes bh
        vv[t]   = v[a0];
    }

    for (int k0 = 0; k0 < E_; k0 += 32) {
        v16h a;
        {
            const v8h h0 = *(const v8h*)(arow + k0 + half * 8);
            const v8h h1 = *(const v8h*)(arow + k0 + 16 + half * 8);
            for (int i = 0; i < 8; ++i) { a[i] = h0[i]; a[8 + i] = h1[i]; }
        }
        for (int t = 0; t < 4; ++t) {
            v16h b;
            const float4 g0 = *(const float4*)(brow[t] + k0 + half * 8);
            const float4 g1 = *(const float4*)(brow[t] + k0 + half * 8 + 4);
            const float4 g2 = *(const float4*)(brow[t] + k0 + half * 8 + 16);
            const float4 g3 = *(const float4*)(brow[t] + k0 + half * 8 + 20);
            b[0] = (_Float16)g0.x; b[1] = (_Float16)g0.y; b[2] = (_Float16)g0.z; b[3] = (_Float16)g0.w;
            b[4] = (_Float16)g1.x; b[5] = (_Float16)g1.y; b[6] = (_Float16)g1.z; b[7] = (_Float16)g1.w;
            b[8] = (_Float16)g2.x; b[9] = (_Float16)g2.y; b[10] = (_Float16)g2.z; b[11] = (_Float16)g2.w;
            b[12] = (_Float16)g3.x; b[13] = (_Float16)g3.y; b[14] = (_Float16)g3.z; b[15] = (_Float16)g3.w;
            acc[t] = __builtin_amdgcn_wmma_f32_16x16x32_f16(
                false, a, false, b, (short)0, acc[t], false, false);
        }
    }

    // Epilogue: rp[r] = sum over this wave's a-columns of v*tanh(qh + d)
    float rp[8] = {0.f, 0.f, 0.f, 0.f, 0.f, 0.f, 0.f, 0.f};
    for (int t = 0; t < 4; ++t)
        for (int r = 0; r < 8; ++r)
            rp[r] += vv[t] * tanhf(qv[t] + acc[t][r]);

    // reduce over the 16 lanes of each half (xor masks stay within half)
    for (int m = 1; m <= 8; m <<= 1)
        for (int r = 0; r < 8; ++r)
            rp[r] += __shfl_xor(rp[r], m, 32);

    if (lrow == 0)
        for (int r = 0; r < 8; ++r)
            red[wave][r + half * 8] = rp[r];
    __syncthreads();

    if (threadIdx.x < 16) {
        float s = 0.f;
        for (int w = 0; w < 8; ++w) s += red[w][threadIdx.x];
        score[m0 + threadIdx.x] = s;
    }
}

// ---------------------------------------------------------------------------
// Masked softmax over L per batch row; writes weights into d_out.
// ---------------------------------------------------------------------------
__global__ __launch_bounds__(256) void k_softmax(const float* __restrict__ score,
                                                 const int* __restrict__ src_lens,
                                                 float* __restrict__ weights) {
    __shared__ float sm[256];
    const int n = blockIdx.x, t = threadIdx.x;
    const int len = src_lens[n];

    float mx = -3.0e38f;
    for (int l = t; l < L_; l += 256) {
        float s = (l < len) ? score[n * L_ + l] : -1.0e10f;
        mx = fmaxf(mx, s);
    }
    sm[t] = mx; __syncthreads();
    for (int o = 128; o > 0; o >>= 1) {
        if (t < o) sm[t] = fmaxf(sm[t], sm[t + o]);
        __syncthreads();
    }
    mx = sm[0]; __syncthreads();

    float sum = 0.f;
    for (int l = t; l < L_; l += 256) {
        float s = (l < len) ? score[n * L_ + l] : -1.0e10f;
        sum += expf(s - mx);
    }
    sm[t] = sum; __syncthreads();
    for (int o = 128; o > 0; o >>= 1) {
        if (t < o) sm[t] += sm[t + o];
        __syncthreads();
    }
    const float inv = 1.0f / sm[0];

    for (int l = t; l < L_; l += 256) {
        float s = (l < len) ? score[n * L_ + l] : -1.0e10f;
        weights[n * L_ + l] = expf(s - mx) * inv;
    }
}

// ---------------------------------------------------------------------------
// ctx[n,e] = sum_l weights[n,l] * ke[n,l,e]   (memory-bound f16 ke sweep)
// ---------------------------------------------------------------------------
__global__ __launch_bounds__(256) void k_ctx(const _Float16* __restrict__ ke,
                                             const float* __restrict__ weights,
                                             float* __restrict__ ctx) {
    __shared__ float wl[256];
    const int n = blockIdx.x >> 1;
    const int e = ((blockIdx.x & 1) * 256) + threadIdx.x;
    float acc = 0.f;
    for (int l0 = 0; l0 < L_; l0 += 256) {
        wl[threadIdx.x] = weights[n * L_ + l0 + threadIdx.x];
        __syncthreads();
        const _Float16* kp = ke + ((size_t)n * L_ + l0) * E_ + e;
        for (int j = 0; j < 256; ++j)
            acc += wl[j] * (float)kp[(size_t)j * E_];
        __syncthreads();
    }
    ctx[n * E_ + e] = acc;
}

// ---------------------------------------------------------------------------
extern "C" void kernel_launch(void* const* d_in, const int* in_sizes, int n_in,
                              void* d_out, int out_size, void* d_ws, size_t ws_size,
                              hipStream_t stream) {
    const float* h_dec    = (const float*)d_in[0];
    const float* h_enc    = (const float*)d_in[1];
    const int*   src_lens = (const int*)d_in[2];
    const float* Wq       = (const float*)d_in[3];
    const float* bq       = (const float*)d_in[4];
    const float* Wk       = (const float*)d_in[5];
    const float* bk       = (const float*)d_in[6];
    const float* Wh       = (const float*)d_in[7];
    const float* bh       = (const float*)d_in[8];
    const float* v        = (const float*)d_in[9];

    float* out     = (float*)d_out;
    float* ctx     = out;                 // N*E
    float* weights = out + N_ * E_;       // N*L

    char* ws = (char*)d_ws;
    _Float16* ke = (_Float16*)ws;                               // N*L*E f16
    float* score = (float*)(ws + (size_t)N_ * L_ * E_ * 2);     // N*L
    float* qe    = score + (size_t)N_ * L_;                     // N*E
    float* qh    = qe + (size_t)N_ * E_;                        // N*A

    k_qe<<<(N_ * E_) / 256, 256, 0, stream>>>(h_dec, Wq, bq, qe);
    k_qh<<<(N_ * A_) / 256, 256, 0, stream>>>(qe, Wh, bh, qh);
    k_ke<<<(N_ * L_) / 64, 256, 0, stream>>>(h_enc, Wk, bk, ke);
    k_score<<<(N_ * L_) / 16, 256, 0, stream>>>(ke, Wh, qh, v, score);
    k_softmax<<<N_, 256, 0, stream>>>(score, src_lens, weights);
    k_ctx<<<N_ * 2, 256, 0, stream>>>(ke, weights, ctx);
}
